// GraphAttentionLayer_57509612093891
// MI455X (gfx1250) — matrix operand order repaired and checked by reading, not compiled, not run on hardware
//
#include <hip/hip_runtime.h>
#include <math.h>

// MI455X / gfx1250 fused GraphAttention layer.
//   K1: Wh = h @ W  (f16 WMMA, f32 accum) + row 1/max(||.||,eps); stores Wh in
//       f16 row-major AND transposed so K2 staging is contiguous tiles.
//   K2: flash-style fused  S = Whn@Whn^T -> mask(adj_eye, n<75 zero-rows)
//       -> online softmax -> O += P @ Wh  -> elu(O / l).
// Heavy math: v_wmma_f32_16x16x32_f16.  Column-tile staging: Tensor Data
// Mover (tensor_load_to_lds) with LDS padding that reproduces the 136-half
// bank-conflict-free pitch in hardware; synced via s_wait_tensorcnt.

typedef __attribute__((ext_vector_type(16))) _Float16 v16h;
typedef __attribute__((ext_vector_type(8)))  _Float16 v8h;
typedef __attribute__((ext_vector_type(8)))  float    v8f;
typedef __attribute__((ext_vector_type(4)))  unsigned int v4u;
typedef __attribute__((ext_vector_type(8)))  unsigned int v8u;

#define NB     8
#define NN     2048
#define DIN    256
#define DOUTC  128
#define ROWZ   75
#define NEGBIG (-9.0e15f)
#define EPSN   1e-12f

#define PITCH2 136   // half-elem pitch (272 B row stride = 64+4 DWORDs; matches TDM pad)
#define PITCH1 264   // half-elem pitch for W^T tile in kernel 1
#define CT     128   // column tile of the fused attention loop

// 16-bit A-matrix fragment (16x32): half 0 -> K = kb+{0..7} and kb+16+{0..7};
// half 1 -> shifted by 8.  Two 16-byte LDS loads.
static __device__ __forceinline__ v16h make_afrag(const _Float16* rowbase, int kb, int half) {
  const v8h lo = *(const v8h*)(rowbase + kb + 8 * half);
  const v8h hi = *(const v8h*)(rowbase + kb + 16 + 8 * half);
  v16h a;
#pragma unroll
  for (int e = 0; e < 8; ++e) { a[e] = lo[e]; a[8 + e] = hi[e]; }
  return a;
}

// 16-bit B-matrix fragment (32x16): half 0 -> K = kb+{0..7} and kb+16+{0..7}; half 1 -> shifted by 8.
static __device__ __forceinline__ v16h make_bfrag(const _Float16* colbase, int kb, int half) {
  const v8h lo = *(const v8h*)(colbase + kb + 8 * half);
  const v8h hi = *(const v8h*)(colbase + kb + 16 + 8 * half);
  v16h b;
#pragma unroll
  for (int e = 0; e < 8; ++e) { b[e] = lo[e]; b[8 + e] = hi[e]; }
  return b;
}

// ---- Tensor Data Mover: 2-D f16 tile (128 rows x 128 halves) -> LDS with
// 4-DWORD row padding (272 B LDS pitch).  D# packing per CDNA5 ISA §8.3/8.4.
static __device__ __forceinline__ void tdm_load_tile_f16_128x128(
    unsigned lds_byte_addr, const void* gtile,
    unsigned tensor_d0, unsigned tensor_d1, unsigned long long stride0) {
  const unsigned long long ga = (unsigned long long)(uintptr_t)gtile;
  v4u g0;
  g0[0] = 1u;                                            // count=1, user load
  g0[1] = lds_byte_addr;                                 // lds_addr
  g0[2] = (unsigned)(ga & 0xffffffffu);                  // global_addr[31:0]
  g0[3] = (unsigned)((ga >> 32) & 0x01ffffffu)           // global_addr[56:32]
        | (2u << 30);                                    // type = 2 ("image")
  v8u g1;
  g1[0] = (1u << 16)        // data_size = 2 bytes
        | (1u << 20)        // pad_enable
        | (5u << 22)        // pad_interval: 64 DWORDs (one 128-half row)
        | (3u << 25);       // pad_amount: 4 DWORDs -> 272 B row pitch
  g1[1] = (tensor_d0 & 0xffffu) << 16;                   // tensor_dim0[15:0]
  g1[2] = (tensor_d0 >> 16) | ((tensor_d1 & 0xffffu) << 16);
  g1[3] = (tensor_d1 >> 16) | (128u << 16);              // tile_dim0 = 128
  g1[4] = 128u;                                          // tile_dim1=128, tile_dim2=0
  g1[5] = (unsigned)(stride0 & 0xffffffffu);             // tensor_dim0_stride
  g1[6] = (unsigned)(stride0 >> 32);                     // (stride1 = 0, 2-D)
  g1[7] = 0u;
  const v4u gz = {0u, 0u, 0u, 0u};                       // groups 2/3: unused dims
  asm volatile("tensor_load_to_lds %0, %1, %2, %3"
               :: "s"(g0), "s"(g1), "s"(gz), "s"(gz)
               : "memory");
}

// ---------------- Kernel 1: Wh = h @ W, row norms ----------------
// 256 threads = 8 waves; wave w computes rows [rb, rb+16) x all 128 cols.
__global__ __launch_bounds__(256) void gat_wh_f16(const float* __restrict__ h,
                                                  const float* __restrict__ W,
                                                  _Float16* __restrict__ WhH,
                                                  _Float16* __restrict__ WhT,
                                                  float* __restrict__ rnorm) {
  __shared__ _Float16 WtH[DOUTC * PITCH1];  // W^T as f16: [c][k]

  const int tid  = threadIdx.x;
  const int w    = tid >> 5;
  const int lane = tid & 31;
  const int half = lane >> 4;
  const int lc   = lane & 15;
  const int bb   = blockIdx.x >> 4;        // / (NN/128)
  const int nblk = blockIdx.x & 15;
  const int rb   = nblk * 128 + w * 16;

  // stage W^T in f16 (coalesced f32 reads, scattered b16 LDS writes; done once)
  for (int idx = tid; idx < DIN * DOUTC; idx += 256) {
    const int k = idx >> 7, c = idx & 127;
    WtH[c * PITCH1 + k] = (_Float16)W[k * DOUTC + c];
  }
  __syncthreads();

  v8f acc[8];
#pragma unroll
  for (int j = 0; j < 8; ++j)
#pragma unroll
    for (int e = 0; e < 8; ++e) acc[j][e] = 0.0f;

  const float* hrow = h + ((size_t)bb * NN + rb + lc) * DIN;
  for (int kb = 0; kb < DIN; kb += 32) {
    const float4 f0 = *(const float4*)(hrow + kb + 8 * half);
    const float4 f1 = *(const float4*)(hrow + kb + 8 * half + 4);
    const float4 f2 = *(const float4*)(hrow + kb + 16 + 8 * half);
    const float4 f3 = *(const float4*)(hrow + kb + 16 + 8 * half + 4);
    v16h a;
    a[0]  = (_Float16)f0.x; a[1]  = (_Float16)f0.y; a[2]  = (_Float16)f0.z; a[3]  = (_Float16)f0.w;
    a[4]  = (_Float16)f1.x; a[5]  = (_Float16)f1.y; a[6]  = (_Float16)f1.z; a[7]  = (_Float16)f1.w;
    a[8]  = (_Float16)f2.x; a[9]  = (_Float16)f2.y; a[10] = (_Float16)f2.z; a[11] = (_Float16)f2.w;
    a[12] = (_Float16)f3.x; a[13] = (_Float16)f3.y; a[14] = (_Float16)f3.z; a[15] = (_Float16)f3.w;
#pragma unroll
    for (int j = 0; j < 8; ++j) {
      const v16h b = make_bfrag(&WtH[(16 * j + lc) * PITCH1], kb, half);
      acc[j] = __builtin_amdgcn_wmma_f32_16x16x32_f16(false, a, false, b,
                                                      (short)0, acc[j], false, false);
    }
  }

  // row sums of squares (C layout: lane holds row = 8*half+v, col = 16*j+lc)
  float ss[8];
#pragma unroll
  for (int v = 0; v < 8; ++v) ss[v] = 0.0f;
#pragma unroll
  for (int j = 0; j < 8; ++j)
#pragma unroll
    for (int v = 0; v < 8; ++v) { const float x = acc[j][v]; ss[v] = fmaf(x, x, ss[v]); }
#pragma unroll
  for (int v = 0; v < 8; ++v) {
    float s = ss[v];
    s += __shfl_xor(s, 1, 32);
    s += __shfl_xor(s, 2, 32);
    s += __shfl_xor(s, 4, 32);
    s += __shfl_xor(s, 8, 32);
    ss[v] = s;
  }

  const int rowg = rb + 8 * half;
#pragma unroll
  for (int j = 0; j < 8; ++j)
#pragma unroll
    for (int v = 0; v < 8; ++v) {
      const _Float16 hv = (_Float16)acc[j][v];
      WhH[((size_t)bb * NN + rowg + v) * DOUTC + 16 * j + lc] = hv;
      WhT[((size_t)bb * DOUTC + 16 * j + lc) * NN + rowg + v] = hv;
    }
  if (lc == 0) {
#pragma unroll
    for (int v = 0; v < 8; ++v) {
      const float nrm = fmaxf(sqrtf(ss[v]), EPSN);
      rnorm[(size_t)bb * NN + rowg + v] = 1.0f / nrm;
    }
  }
}

// ---------------- Kernel 2: fused attention ----------------
// 128 threads = 4 waves; workgroup owns 64 rows, streams 128-column tiles.
__global__ __launch_bounds__(128) void gat_attn_f16(const int* __restrict__ adjE,
                                                    const _Float16* __restrict__ WhH,
                                                    const _Float16* __restrict__ WhT,
                                                    const float* __restrict__ rnorm,
                                                    float* __restrict__ out) {
  __shared__ _Float16 rowWhn[64 * PITCH2];      // normalized row block [r][d]
  __shared__ _Float16 colWh [CT * PITCH2];      // column tile [m][d]   (S B-op)
  __shared__ _Float16 colWhT[DOUTC * PITCH2];   // column tile [d][m]   (O B-op)
  __shared__ _Float16 Pbuf  [64 * PITCH2];      // per-wave softmax probs

  const int tid  = threadIdx.x;
  const int w    = tid >> 5;
  const int lane = tid & 31;
  const int half = lane >> 4;
  const int lc   = lane & 15;
  const int bb   = blockIdx.x >> 5;             // / (NN/64)
  const int nblk = blockIdx.x & 31;
  const int rbase = nblk * 64;
  const size_t whBase = (size_t)bb * NN * DOUTC;

  // stage normalized rows: Whn = Wh * rnorm (f32 multiply, f16 store)
  for (int idx = tid; idx < 64 * 16; idx += 128) {
    const int r = idx >> 4, c8 = (idx & 15) << 3;
    const float rn = rnorm[(size_t)bb * NN + rbase + r];
    const v8h src = *(const v8h*)(WhH + whBase + (size_t)(rbase + r) * DOUTC + c8);
    v8h dstv;
#pragma unroll
    for (int e = 0; e < 8; ++e) dstv[e] = (_Float16)((float)src[e] * rn);
    *(v8h*)&rowWhn[r * PITCH2 + c8] = dstv;
  }

  float rmax[8], rsum[8];
  v8f oacc[8];
#pragma unroll
  for (int v = 0; v < 8; ++v) { rmax[v] = -3.402823466e38f; rsum[v] = 0.0f; }
#pragma unroll
  for (int jd = 0; jd < 8; ++jd)
#pragma unroll
    for (int e = 0; e < 8; ++e) oacc[jd][e] = 0.0f;

  const int rowg0 = rbase + w * 16 + 8 * half;  // lane's first output row
  const int* adjRow = adjE + ((size_t)bb * NN + rowg0) * NN;

  __syncthreads();

  for (int ct = 0; ct < NN; ct += CT) {
    if (ct) __syncthreads();                    // prior tile fully consumed
    // Tensor Data Mover stages both column tiles (one DMA each); TENSORcnt is
    // per-wave, so wave 0 issues + waits, the barrier releases everyone else.
    if (w == 0) {
      tdm_load_tile_f16_128x128((unsigned)(uintptr_t)&colWh[0],
                                WhH + whBase + (size_t)ct * DOUTC,
                                DOUTC, NN, DOUTC);
      tdm_load_tile_f16_128x128((unsigned)(uintptr_t)&colWhT[0],
                                WhT + (size_t)bb * DOUTC * NN + ct,
                                NN, DOUTC, NN);
      __builtin_amdgcn_s_wait_tensorcnt(0);
    }
    __syncthreads();

    float rnreg[8];                              // column 1/norm (post-scale)
#pragma unroll
    for (int j = 0; j < 8; ++j) rnreg[j] = rnorm[(size_t)bb * NN + ct + 16 * j + lc];

    if (ct + CT < NN) __builtin_prefetch(adjRow + ct + CT + (lane << 2), 0, 0);

    // S = rowWhn @ colWh^T  (K = 128 features)
    v8f sacc[8];
#pragma unroll
    for (int j = 0; j < 8; ++j)
#pragma unroll
      for (int e = 0; e < 8; ++e) sacc[j][e] = 0.0f;
    const _Float16* arow = &rowWhn[(w * 16 + lc) * PITCH2];
#pragma unroll
    for (int kb = 0; kb < DOUTC; kb += 32) {
      const v16h a = make_afrag(arow, kb, half);
#pragma unroll
      for (int j = 0; j < 8; ++j) {
        const v16h b = make_bfrag(&colWh[(16 * j + lc) * PITCH2], kb, half);
        sacc[j] = __builtin_amdgcn_wmma_f32_16x16x32_f16(false, a, false, b,
                                                         (short)0, sacc[j], false, false);
      }
    }

    // normalize columns, zero rows n<75, apply adjacency mask
#pragma unroll
    for (int j = 0; j < 8; ++j) {
      const int m = ct + 16 * j + lc;
#pragma unroll
      for (int v = 0; v < 8; ++v) {
        float s = sacc[j][v] * rnreg[j];
        if (rowg0 + v < ROWZ) s = 0.0f;
        const int av = adjRow[(size_t)v * NN + m];
        sacc[j][v] = (av > 0) ? s : NEGBIG;
      }
    }

    // online softmax per row (cross-lane reduce over the 16 lanes of a row)
#pragma unroll
    for (int v = 0; v < 8; ++v) {
      float tmax = -3.402823466e38f;
#pragma unroll
      for (int j = 0; j < 8; ++j) tmax = fmaxf(tmax, sacc[j][v]);
      tmax = fmaxf(tmax, __shfl_xor(tmax, 1, 32));
      tmax = fmaxf(tmax, __shfl_xor(tmax, 2, 32));
      tmax = fmaxf(tmax, __shfl_xor(tmax, 4, 32));
      tmax = fmaxf(tmax, __shfl_xor(tmax, 8, 32));
      const float mnew = fmaxf(rmax[v], tmax);
      const float al = __expf(rmax[v] - mnew);
      rmax[v] = mnew;
      float ps = 0.0f;
#pragma unroll
      for (int j = 0; j < 8; ++j) {
        const float p = __expf(sacc[j][v] - mnew);
        sacc[j][v] = p;
        ps += p;
      }
      ps += __shfl_xor(ps, 1, 32);
      ps += __shfl_xor(ps, 2, 32);
      ps += __shfl_xor(ps, 4, 32);
      ps += __shfl_xor(ps, 8, 32);
      rsum[v] = rsum[v] * al + ps;
#pragma unroll
      for (int jd = 0; jd < 8; ++jd) oacc[jd][v] *= al;
    }

    // park P in LDS (per-wave region; same-wave DS ordering only)
#pragma unroll
    for (int j = 0; j < 8; ++j)
#pragma unroll
      for (int v = 0; v < 8; ++v)
        Pbuf[(w * 16 + 8 * half + v) * PITCH2 + 16 * j + lc] = (_Float16)sacc[j][v];
    asm volatile("s_wait_dscnt 0" ::: "memory");

    // O += P @ Wh_cols  (K = 128 neighbors of this tile)
    const _Float16* prow = &Pbuf[(w * 16 + lc) * PITCH2];
#pragma unroll
    for (int kb = 0; kb < CT; kb += 32) {
      const v16h a = make_afrag(prow, kb, half);
#pragma unroll
      for (int jd = 0; jd < 8; ++jd) {
        const v16h b = make_bfrag(&colWhT[(16 * jd + lc) * PITCH2], kb, half);
        oacc[jd] = __builtin_amdgcn_wmma_f32_16x16x32_f16(false, a, false, b,
                                                          (short)0, oacc[jd], false, false);
      }
    }
  }

  // epilogue: divide by softmax denominator, ELU, store f32
#pragma unroll
  for (int v = 0; v < 8; ++v) {
    const float inv = 1.0f / rsum[v];
#pragma unroll
    for (int jd = 0; jd < 8; ++jd) {
      const float x = oacc[jd][v] * inv;
      out[((size_t)bb * NN + rowg0 + v) * DOUTC + 16 * jd + lc] =
          (x > 0.0f) ? x : expm1f(x);
    }
  }
}

extern "C" void kernel_launch(void* const* d_in, const int* in_sizes, int n_in,
                              void* d_out, int out_size, void* d_ws, size_t ws_size,
                              hipStream_t stream) {
  (void)in_sizes; (void)n_in; (void)out_size; (void)ws_size;
  const float* h    = (const float*)d_in[0];
  // d_in[1] = adj : unused by the reference math (only adj_eye masks)
  const int*   adjE = (const int*)d_in[2];
  const float* W    = (const float*)d_in[3];
  float* out = (float*)d_out;

  // workspace: Wh f16 (4 MB) + Wh^T f16 (4 MB) + rnorm f32 (64 KB)
  _Float16* WhH = (_Float16*)d_ws;
  _Float16* WhT = WhH + (size_t)NB * NN * DOUTC;
  float* rnorm  = (float*)(WhT + (size_t)NB * NN * DOUTC);

  gat_wh_f16<<<dim3(NB * (NN / 128)), dim3(256), 0, stream>>>(h, W, WhH, WhT, rnorm);
  gat_attn_f16<<<dim3(NB * (NN / 64)), dim3(128), 0, stream>>>(adjE, WhH, WhT, rnorm, out);
}
